// SignatureBlock_83691732729936
// MI455X (gfx1250) — compile-verified, matrix operand order checked by reading, not run on hardware
//
#include <hip/hip_runtime.h>
#include <hip/hip_bf16.h>

typedef __attribute__((ext_vector_type(16))) _Float16 v16h;
typedef __attribute__((ext_vector_type(8)))  _Float16 v8h;
typedef __attribute__((ext_vector_type(8)))  float    v8f;

#define T_ROWS 254      // aug rows
#define KT     253      // effective K for s2 (nonzero lead-lag increments)
#define CH     145      // signature channel count (E+1+128)
#define CHP    160      // padded to 10 WMMA tiles
#define SIGC   21170    // 145 + 145*145
#define CP     21184    // sig K padded to multiple of 32
#define OUTC   256
#define NB     256      // B*H

// ---------------------------------------------------------------------------
// Kernel 1: per-n fused conv1 -> conv2 -> aug -> signature (s1 + s2 via WMMA)
// One block per n (256 blocks), 256 threads = 8 waves.
// s2 = M^T D with M(t)=0.5*(aug[t]+aug[t+1])-aug[0], D(t)=aug[t+1]-aug[t].
// ---------------------------------------------------------------------------
__global__ __launch_bounds__(256) void k1_sig(
    const float* __restrict__ q,
    const float* __restrict__ w1, const float* __restrict__ b1,
    const float* __restrict__ w2, const float* __restrict__ b2,
    _Float16* __restrict__ sig)              // [NB][CP] fp16
{
  __shared__ float    augC[33 * CHP];        // current 33 aug rows (f32)
  __shared__ float    aug0[CHP];             // aug row 0 (persisted)
  __shared__ float    c1buf[33 * 64];        // conv1 outputs for chunk
  __shared__ _Float16 Mt[CHP * 32];          // M^T chunk: [c][k], 64B rows
  __shared__ _Float16 Dt[CHP * 32];          // D^T chunk: [c][k]

  const int n    = blockIdx.x;
  const int b    = n >> 3;
  const int h    = n & 7;
  const int tid  = threadIdx.x;
  const int wave = tid >> 5;                 // 0..7 (wave32)
  const int lane = tid & 31;
  const int lrow = lane & 15;
  const int lhalf= lane >> 4;

  // x[n][l][e] = q[b][l][h][e]; row stride = H*E = 128 floats
  const float* qb = q + ((size_t)b * 256 * 8 + h) * 16;

  v8f acc[13] = {};                          // up to 13 16x16 s2 tiles / wave

  for (int kc = 0; kc < 8; ++kc) {
    const int r0    = kc * 32;
    const int nrows = (kc == 7) ? 30 : 33;   // rows r0 .. r0+nrows-1 (<=253)

    // ---- conv1 (3x16 -> 64, bias, no relu) ----
    for (int task = tid; task < nrows * 64; task += 256) {
      const int rl = task >> 6, co = task & 63;
      const int tg = r0 + rl;
      float a = b1[co];
      #pragma unroll
      for (int w = 0; w < 3; ++w) {
        const float* xr = qb + (size_t)(tg + w) * 128;
        #pragma unroll
        for (int ci = 0; ci < 16; ++ci)
          a += xr[ci] * w1[(w * 16 + ci) * 64 + co];
      }
      c1buf[rl * 64 + co] = a;
    }
    __syncthreads();

    // ---- assemble aug rows: [trunc(16) | time(1) | relu(conv2)(128) | pad] ----
    for (int task = tid; task < nrows * CHP; task += 256) {
      const int rl = task / CHP, c = task % CHP;
      const int tg = r0 + rl;
      float v;
      if (c < 16) {
        v = qb[(size_t)(tg + 2) * 128 + c];              // trunc = x[:, t+2]
      } else if (c == 16) {
        v = (float)tg * (1.0f / 253.0f);                 // time channel
      } else if (c < CH) {
        const int co = c - 17;
        float a = b2[co];
        const float* c1 = &c1buf[rl * 64];
        for (int ci = 0; ci < 64; ++ci)
          a += c1[ci] * w2[ci * 128 + co];
        v = a > 0.0f ? a : 0.0f;                         // relu
      } else {
        v = 0.0f;                                        // channel pad
      }
      augC[rl * CHP + c] = v;
      if (kc == 0 && rl == 0) aug0[c] = v;
    }
    __syncthreads();

    // ---- build transposed f16 M,D fragments for this K-chunk ----
    for (int task = tid; task < CHP * 32; task += 256) {
      const int c = task >> 5, k = task & 31;
      const int t = r0 + k;
      float dv = 0.0f, mv = 0.0f;
      if (t < KT) {
        const float a0 = augC[k * CHP + c];
        const float a1 = augC[(k + 1) * CHP + c];
        dv = a1 - a0;
        mv = 0.5f * (a0 + a1) - aug0[c];
      }
      Dt[c * 32 + k] = (_Float16)dv;
      Mt[c * 32 + k] = (_Float16)mv;
    }
    __syncthreads();

    // ---- WMMA: accumulate the 10x10 tile grid of S2 += M^T D ----
    #pragma unroll
    for (int u = 0; u < 13; ++u) {
      const int tt = wave + 8 * u;                       // wave-uniform
      if (tt < 100) {
        const int it = tt / 10, jt = tt % 10;
        const _Float16* mrow = &Mt[(it * 16 + lrow) * 32];
        const _Float16* drow = &Dt[(jt * 16 + lrow) * 32];
        // A[m][K], lane=m+16*h holds K in {8h..8h+7} and {16+8h..23+8h}
        v8h alo = *(const v8h*)(mrow + lhalf * 8);
        v8h ahi = *(const v8h*)(mrow + 16 + lhalf * 8);
        v8h blo = *(const v8h*)(drow + lhalf * 8);
        v8h bhi = *(const v8h*)(drow + 16 + lhalf * 8);
        v16h av, bv;
        #pragma unroll
        for (int e = 0; e < 8; ++e) {
          av[e] = alo[e]; av[8 + e] = ahi[e];
          bv[e] = blo[e]; bv[8 + e] = bhi[e];
        }
        acc[u] = __builtin_amdgcn_wmma_f32_16x16x32_f16(
            false, av, false, bv, (short)0, acc[u], false, false);
      }
    }
    __syncthreads();
  }

  // ---- s1 = aug[253] - aug[0]  (row 253 is local row 29 of last chunk) ----
  if (tid < CH)
    sig[(size_t)n * CP + tid] = (_Float16)(augC[29 * CHP + tid] - aug0[tid]);
  for (int i = tid; i < CP - SIGC; i += 256)             // zero K-pad
    sig[(size_t)n * CP + SIGC + i] = (_Float16)0.0f;

  // ---- scatter s2 tiles (C/D layout: lane=n+16*(m>=8), vgpr r = m%8) ----
  #pragma unroll
  for (int u = 0; u < 13; ++u) {
    const int tt = wave + 8 * u;
    if (tt < 100) {
      const int it = tt / 10, jt = tt % 10;
      const int j = jt * 16 + lrow;
      #pragma unroll
      for (int r = 0; r < 8; ++r) {
        const int i = it * 16 + r + 8 * lhalf;
        if (i < CH && j < CH)
          sig[(size_t)n * CP + CH + (size_t)i * CH + j] = (_Float16)acc[u][r];
      }
    }
  }
}

// ---------------------------------------------------------------------------
// Kernel 2: lin_w fp32 -> fp16 with zero rows in the K-pad region
// ---------------------------------------------------------------------------
__global__ __launch_bounds__(256) void k2_cvtw(
    const float* __restrict__ lw, _Float16* __restrict__ lwh)
{
  const size_t i = (size_t)blockIdx.x * 256 + threadIdx.x;
  if (i >= (size_t)CP * OUTC) return;
  const int cp = (int)(i >> 8);
  const int o  = (int)(i & 255);
  const float v = (cp < SIGC) ? lw[(size_t)cp * OUTC + o] : 0.0f;
  lwh[i] = (_Float16)v;
}

// ---------------------------------------------------------------------------
// Kernel 3: out(256x256) = sig(256xCP) @ lin_w(CPx256) + lin_b  (f16 WMMA)
// 64 blocks x 4 waves; block owns 4 i-tiles x 1 j-tile; B staged in LDS.
// ---------------------------------------------------------------------------
__global__ __launch_bounds__(128) void k3_gemm(
    const _Float16* __restrict__ sig, const _Float16* __restrict__ lwh,
    const float* __restrict__ lb, float* __restrict__ out)
{
  __shared__ _Float16 Bt[16 * 40];           // B^T tile: [n][k], 80B rows (16B aligned)

  const int tid  = threadIdx.x;
  const int wave = tid >> 5;
  const int lane = tid & 31;
  const int lrow = lane & 15;
  const int lhalf= lane >> 4;
  const int jt   = blockIdx.x & 15;
  const int it   = (blockIdx.x >> 4) * 4 + wave;
  const int arow = it * 16 + lrow;           // sig row this lane reads

  v8f acc = {};
  for (int k0 = 0; k0 < CP; k0 += 32) {
    // stage + transpose 32x16 B tile into LDS
    for (int idx = tid; idx < 512; idx += 128) {
      const int k = idx >> 4, nn = idx & 15;
      Bt[nn * 40 + k] = lwh[(size_t)(k0 + k) * OUTC + jt * 16 + nn];
    }
    if (k0 + 32 < CP)                        // prefetch next B tile region
      __builtin_prefetch(&lwh[(size_t)(k0 + 32) * OUTC + jt * 16], 0, 3);
    __syncthreads();

    const _Float16* ap = sig + (size_t)arow * CP + k0;
    v8h alo = *(const v8h*)(ap + lhalf * 8);
    v8h ahi = *(const v8h*)(ap + 16 + lhalf * 8);
    const _Float16* bp = &Bt[lrow * 40];
    v8h blo = *(const v8h*)(bp + lhalf * 8);
    v8h bhi = *(const v8h*)(bp + 16 + lhalf * 8);
    v16h av, bv;
    #pragma unroll
    for (int e = 0; e < 8; ++e) {
      av[e] = alo[e]; av[8 + e] = ahi[e];
      bv[e] = blo[e]; bv[8 + e] = bhi[e];
    }
    acc = __builtin_amdgcn_wmma_f32_16x16x32_f16(
        false, av, false, bv, (short)0, acc, false, false);
    __syncthreads();
  }

  const int col  = jt * 16 + lrow;
  const float bias = lb[col];
  #pragma unroll
  for (int r = 0; r < 8; ++r) {
    const int m = it * 16 + r + 8 * lhalf;
    out[(size_t)m * OUTC + col] = acc[r] + bias;
  }
}

// ---------------------------------------------------------------------------
extern "C" void kernel_launch(void* const* d_in, const int* in_sizes, int n_in,
                              void* d_out, int out_size, void* d_ws, size_t ws_size,
                              hipStream_t stream) {
  const float* q  = (const float*)d_in[0];
  // d_in[1]=k, d_in[2]=v, d_in[3]=mask : unused by the reference
  const float* w1 = (const float*)d_in[4];   // conv1_w (3,16,64)
  const float* b1 = (const float*)d_in[5];   // conv1_b (64)
  const float* w2 = (const float*)d_in[6];   // conv2_w (1,64,128)
  const float* b2 = (const float*)d_in[7];   // conv2_b (128)
  const float* lw = (const float*)d_in[8];   // lin_w (21170,256)
  const float* lb = (const float*)d_in[9];   // lin_b (256)
  float* out = (float*)d_out;

  _Float16* sig = (_Float16*)d_ws;                       // NB*CP fp16 (10.8 MB)
  _Float16* lwh = sig + (size_t)NB * CP;                 // CP*256 fp16 (10.8 MB)

  k2_cvtw<<<CP, 256, 0, stream>>>(lw, lwh);
  k1_sig<<<NB, 256, 0, stream>>>(q, w1, b1, w2, b2, sig);
  k3_gemm<<<64, 128, 0, stream>>>(sig, lwh, lb, out);
}
